// Model_15908558865502
// MI455X (gfx1250) — matrix operand order repaired and checked by reading, not compiled, not run on hardware
//
#include <hip/hip_runtime.h>
#include <hip/hip_bf16.h>
#include <stdint.h>

typedef __attribute__((ext_vector_type(2)))  float  v2f;
typedef __attribute__((ext_vector_type(8)))  float  v8f;
typedef __attribute__((ext_vector_type(8)))  __bf16 v8bf;
typedef __attribute__((ext_vector_type(16))) __bf16 v16bf;

#define NB   2
#define NH   64
#define SEQ  1024
#define DIM  1024
#define DH   16

// ---------------------------------------------------------------------------
// f32 -> bf16 (round to nearest even)
// ---------------------------------------------------------------------------
__global__ __launch_bounds__(256) void cvt_bf16(const float* __restrict__ in,
                                                unsigned short* __restrict__ out, int n) {
  int i = blockIdx.x * 256 + threadIdx.x;
  if (i < n) {
    unsigned u = __float_as_uint(in[i]);
    unsigned r = (u + 0x7FFFu + ((u >> 16) & 1u)) >> 16;
    out[i] = (unsigned short)r;
  }
}

// ---------------------------------------------------------------------------
// QKV projection: y = x @ W^T, stored head-split [B,H,S,DH] in f32.
// One wave = one 16x16 output tile; K loop in steps of 32 with
// v_wmma_f32_16x16x32_bf16. gridDim = (8 n-groups, 128 m-tiles, 3 weights).
// ---------------------------------------------------------------------------
__global__ __launch_bounds__(256) void qkv_gemm(const unsigned short* __restrict__ xb,
                                                const unsigned short* __restrict__ wb,
                                                float* __restrict__ outf) {
  const int wave = threadIdx.x >> 5;
  const int lane = threadIdx.x & 31;
  const int hi   = lane >> 4;
  const int l16  = lane & 15;
  const int m    = blockIdx.y;                 // row tile: rows 16m..16m+15 of [B*S]
  const int nt   = blockIdx.x * 8 + wave;      // col tile == head index
  const int wsel = blockIdx.z;                 // 0=Q 1=K 2=V

  const unsigned short* W  = wb + (size_t)wsel * (DIM * DIM);
  float* Out               = outf + (size_t)wsel * ((size_t)NB * NH * SEQ * DH);

  const unsigned short* aptr = xb + (size_t)(m * 16 + l16) * DIM;   // x row
  const unsigned short* bptr = W  + (size_t)(nt * 16 + l16) * DIM;  // W row (output col)

  v8f acc = {};
#pragma unroll 4
  for (int kk = 0; kk < DIM / 32; ++kk) {
    const int kb = kk * 32;
    // A (16x32): lane group hi selects K half 0..15 / 16..31, contiguous (matches B)
    v8bf a0 = *(const v8bf*)(aptr + kb + 16 * hi);
    v8bf a1 = *(const v8bf*)(aptr + kb + 16 * hi + 8);
    v16bf A = __builtin_shufflevector(a0, a1, 0,1,2,3,4,5,6,7,8,9,10,11,12,13,14,15);
    // B (32x16): lane group hi selects K half 0..15 / 16..31, contiguous in W row
    v16bf Bm = *(const v16bf*)(bptr + kb + 16 * hi);
    acc = __builtin_amdgcn_wmma_f32_16x16x32_bf16(false, A, false, Bm,
                                                  (short)0, acc, false, false);
  }
  // C tile element (M = v + 8*hi, N = l16); scatter into [B,H,S,DH]
#pragma unroll
  for (int v = 0; v < 8; ++v) {
    int rg = m * 16 + v + 8 * hi;                 // global row in [0, B*S)
    int bb = rg >> 10;
    int ss = rg & 1023;
    Out[(((size_t)bb * NH + nt) * SEQ + ss) * DH + l16] = acc[v];
  }
}

// ---------------------------------------------------------------------------
// Exact JAX threefry2x32, key = (0, 42)
// ---------------------------------------------------------------------------
__device__ __forceinline__ unsigned tf_rotl(unsigned x, int r) {
  return (x << r) | (x >> (32 - r));
}
__device__ __forceinline__ void threefry2x32(unsigned x0, unsigned x1,
                                             unsigned& y0, unsigned& y1) {
  const unsigned ks0 = 0u, ks1 = 42u, ks2 = 0u ^ 42u ^ 0x1BD11BDAu;
  x0 += ks0; x1 += ks1;
#define TF_R(r) { x0 += x1; x1 = tf_rotl(x1, r); x1 ^= x0; }
  TF_R(13) TF_R(15) TF_R(26) TF_R(6)   x0 += ks1; x1 += ks2 + 1u;
  TF_R(17) TF_R(29) TF_R(16) TF_R(24)  x0 += ks2; x1 += ks0 + 2u;
  TF_R(13) TF_R(15) TF_R(26) TF_R(6)   x0 += ks0; x1 += ks1 + 3u;
  TF_R(17) TF_R(29) TF_R(16) TF_R(24)  x0 += ks1; x1 += ks2 + 4u;
  TF_R(13) TF_R(15) TF_R(26) TF_R(6)   x0 += ks2; x1 += ks0 + 5u;
#undef TF_R
  y0 = x0; y1 = x1;
}

// ---------------------------------------------------------------------------
// Fused attention. One WG per (head, batch). K,V for the head live entirely
// in LDS (loaded via CDNA5 async-to-LDS). Each wave owns 16 query rows at a
// time: QK^T and P.V with v_wmma_f32_16x16x4_f32, un-shifted exp (logits O(1)),
// exact threefry dropout folded into the PV accumulation.
// ---------------------------------------------------------------------------
__global__ __launch_bounds__(256) void attn_fused(const float* __restrict__ Qf,
                                                  const float* __restrict__ Kf,
                                                  const float* __restrict__ Vf,
                                                  float* __restrict__ Out) {
  __shared__ float Ks[SEQ * DH];      // 64 KB
  __shared__ float Vs[SEQ * DH];      // 64 KB
  __shared__ float Pb[8][16 * 16];    // 8 KB (per-wave P staging)

  const int h    = blockIdx.x;
  const int b    = blockIdx.y;
  const int wave = threadIdx.x >> 5;
  const int lane = threadIdx.x & 31;
  const int hi   = lane >> 4;
  const int l16  = lane & 15;

  const size_t head_off = ((size_t)b * NH + h) * SEQ * DH;
  const float4* Kg = (const float4*)(Kf + head_off);
  const float4* Vg = (const float4*)(Vf + head_off);
  const unsigned ksOff = (unsigned)(uintptr_t)&Ks[0];
  const unsigned vsOff = (unsigned)(uintptr_t)&Vs[0];

  // 4096 float4 per matrix, 16 per thread: async copy global -> LDS
#pragma unroll
  for (int i = 0; i < 16; ++i) {
    int fi = i * 256 + threadIdx.x;
    asm volatile("global_load_async_to_lds_b128 %0, %1, off"
                 :: "v"(ksOff + fi * 16), "v"(Kg + fi) : "memory");
    asm volatile("global_load_async_to_lds_b128 %0, %1, off"
                 :: "v"(vsOff + fi * 16), "v"(Vg + fi) : "memory");
  }
  asm volatile("s_wait_asynccnt 0" ::: "memory");
  __syncthreads();

  for (int qb = wave; qb < SEQ / 16; qb += 8) {
    const int s0 = qb * 16;
    // Q block in f32 A-layout (16x4 chunks): lane: row=l16, K = 4c+2hi+{0,1}
    v2f qa[4];
#pragma unroll
    for (int c = 0; c < 4; ++c)
      qa[c] = *(const v2f*)(Qf + head_off + (size_t)(s0 + l16) * DH + 4 * c + 2 * hi);

    v8f acc = {};
    float ls[8];
#pragma unroll
    for (int v = 0; v < 8; ++v) ls[v] = 0.0f;

    for (int t = 0; t < SEQ / 16; ++t) {
      const int t0 = t * 16;
      // ---- scores = Q . K^T (4 chained K=4 f32 WMMAs) ----
      v8f sc = {};
#pragma unroll
      for (int c = 0; c < 4; ++c) {
        v2f kb = *(const v2f*)&Ks[(t0 + l16) * DH + 4 * c + 2 * hi];
        sc = __builtin_amdgcn_wmma_f32_16x16x4_f32(false, qa[c], false, kb,
                                                   (short)0, sc, false, false);
      }
      // ---- exp + exact-JAX dropout mask ----
      float p[8];
#pragma unroll
      for (int v = 0; v < 8; ++v) {
        float e = __expf(sc[v] * 0.25f);          // 1/sqrt(dh)
        ls[v] += e;                               // unmasked denominator partial
        unsigned row = (unsigned)(s0 + v + 8 * hi);
        unsigned j   = (((unsigned)h << 10) + row) * 1024u + (unsigned)(t0 + l16);
        unsigned y0, y1;
        threefry2x32(j, j + 0x4000000u, y0, y1);  // counters (j, j+N/2)
        unsigned y = b ? y1 : y0;
        float u = __uint_as_float((y >> 9) | 0x3F800000u) - 1.0f;
        p[v] = (u < 0.9f) ? e : 0.0f;
      }
      // ---- C-layout -> A-layout via per-wave LDS tile (DS in-order in wave) ----
#pragma unroll
      for (int v = 0; v < 8; ++v)
        Pb[wave][(v + 8 * hi) * 16 + l16] = p[v];
      asm volatile("" ::: "memory");
      // ---- acc += P . V ----
#pragma unroll
      for (int c = 0; c < 4; ++c) {
        v2f pa = *(const v2f*)&Pb[wave][l16 * 16 + 4 * c + 2 * hi];
        int k0 = t0 + 4 * c + 2 * hi;
        v2f vb;
        vb.x = Vs[k0 * DH + l16];
        vb.y = Vs[(k0 + 1) * DH + l16];
        acc = __builtin_amdgcn_wmma_f32_16x16x4_f32(false, pa, false, vb,
                                                    (short)0, acc, false, false);
      }
    }
    // ---- row-sum reduce (16-lane groups) and normalize: out = acc/(0.9*L) ----
#pragma unroll
    for (int v = 0; v < 8; ++v) {
      float L = ls[v];
      for (int mm = 1; mm < 16; mm <<= 1) L += __shfl_xor(L, mm, 16);
      Out[head_off + (size_t)(s0 + v + 8 * hi) * DH + l16] = acc[v] / (0.9f * L);
    }
  }
}

// ---------------------------------------------------------------------------
extern "C" void kernel_launch(void* const* d_in, const int* in_sizes, int n_in,
                              void* d_out, int out_size, void* d_ws, size_t ws_size,
                              hipStream_t stream) {
  (void)in_sizes; (void)n_in; (void)out_size;
  const float* x1 = (const float*)d_in[0];
  const float* Wq = (const float*)d_in[1];
  const float* Wk = (const float*)d_in[2];
  const float* Wv = (const float*)d_in[3];

  char* ws = (char*)d_ws;
  const size_t MiB = 1u << 20;
  // [0,4M): x_bf16  [4M,6M): Wq_bf16  [6M,8M): Wk_bf16  [8M,10M): Wv_bf16
  // [10M,18M): Q f32  [18M,26M): K f32  [26M,34M): V f32
  unsigned short* xb  = (unsigned short*)(ws);
  unsigned short* wbb = (unsigned short*)(ws + 4 * MiB);   // 3 contiguous weights
  float*          qkv = (float*)(ws + 10 * MiB);           // 3 contiguous tensors
  if (ws_size < 34 * MiB) return;

  const int NX = NB * SEQ * DIM;   // 2,097,152
  const int NW = DIM * DIM;        // 1,048,576
  cvt_bf16<<<(NX + 255) / 256, 256, 0, stream>>>(x1, xb, NX);
  cvt_bf16<<<(NW + 255) / 256, 256, 0, stream>>>(Wq, wbb,          NW);
  cvt_bf16<<<(NW + 255) / 256, 256, 0, stream>>>(Wk, wbb + NW,     NW);
  cvt_bf16<<<(NW + 255) / 256, 256, 0, stream>>>(Wv, wbb + 2 * NW, NW);

  dim3 gg(8, (NB * SEQ) / 16, 3);
  qkv_gemm<<<gg, 256, 0, stream>>>(xb, wbb, qkv);

  const size_t TEN = (size_t)NB * NH * SEQ * DH;           // 2,097,152 elems
  attn_fused<<<dim3(NH, NB), 256, 0, stream>>>(qkv, qkv + TEN, qkv + 2 * TEN,
                                               (float*)d_out);
}
